// MultiHeadAttention_91122026152270
// MI455X (gfx1250) — matrix operand order repaired and checked by reading, not compiled, not run on hardware
//
#include <hip/hip_runtime.h>

typedef __attribute__((ext_vector_type(16))) _Float16 v16h;
typedef __attribute__((ext_vector_type(8)))  _Float16 v8h;
typedef __attribute__((ext_vector_type(8)))  float    v8f;

#define B_  2
#define S_  2048
#define D_  1024
#define H_  16
#define DK_ 64

#define WMMA(a,b,c) (c) = __builtin_amdgcn_wmma_f32_16x16x32_f16( \
    false, (a), false, (b), (short)0, (c), false, false)

static __device__ __forceinline__ v8f zero8() {
  v8f z = {0.f,0.f,0.f,0.f,0.f,0.f,0.f,0.f};
  return z;
}

static __device__ __forceinline__ v16h frag_join(v8h lo, v8h hi) {
  v16h r;
#pragma unroll
  for (int i = 0; i < 8; ++i) { r[i] = lo[i]; r[i + 8] = hi[i]; }
  return r;
}

// ---------------------------------------------------------------------------
// Kernel 1: per-head projection  Out[b,h,s,dk] = X[b,s,:] @ W[h,:,dk]  (f16 out)
// TR=1 stores Out as [B,H,DK,S] (V: makes P@V B-frags contiguous AND the
// epilogue a packed b128 store). TR=0 stores [B,H,S,DK] via LDS transpose.
// ---------------------------------------------------------------------------
template <int TR>
__global__ __launch_bounds__(128)
void proj_kernel(const float* __restrict__ X, const float* __restrict__ W,
                 _Float16* __restrict__ Out)
{
  const int b = blockIdx.z, h = blockIdx.y;
  const int sBase = blockIdx.x * 64;
  const int tid  = threadIdx.x;
  const int lane = tid & 31;
  const int wv   = tid >> 5;          // 0..3, one 16-row tile per wave
  const int bh   = b * H_ + h;
  const int m = lane & 15, g = lane >> 4;

  __shared__ _Float16 lds[5120];
  _Float16* At = lds;                 // 64 rows x 32 k, stride 40
  _Float16* Wt = lds + 64 * 40;       // W tile transposed [n][k], stride 40

  const float* Xb = X + (size_t)b * S_ * D_;
  const float* Wh = W + (size_t)h * D_ * DK_;

  v8f acc[4];
#pragma unroll
  for (int nt = 0; nt < 4; ++nt) acc[nt] = zero8();

  for (int kt = 0; kt < D_ / 32; ++kt) {
    // ---- cooperative stage to LDS ----
    {
      const int row  = tid >> 1;          // 0..63
      const int ksel = (tid & 1) * 16;    // 0 or 16
      const float* src = Xb + (size_t)(sBase + row) * D_ + kt * 32 + ksel;
      v8h ha, hb;
#pragma unroll
      for (int j = 0; j < 8; ++j) ha[j] = (_Float16)src[j];
#pragma unroll
      for (int j = 0; j < 8; ++j) hb[j] = (_Float16)src[8 + j];
      *(v8h*)(At + row * 40 + ksel)     = ha;
      *(v8h*)(At + row * 40 + ksel + 8) = hb;

      const int n = row;                  // output column; loads coalesce across tid
      v8h wa, wb;
#pragma unroll
      for (int j = 0; j < 8; ++j)
        wa[j] = (_Float16)Wh[(size_t)(kt * 32 + ksel + j) * DK_ + n];
#pragma unroll
      for (int j = 0; j < 8; ++j)
        wb[j] = (_Float16)Wh[(size_t)(kt * 32 + ksel + 8 + j) * DK_ + n];
      *(v8h*)(Wt + n * 40 + ksel)     = wa;
      *(v8h*)(Wt + n * 40 + ksel + 8) = wb;
    }
    __syncthreads();

    const _Float16* arow = At + (wv * 16 + m) * 40 + g * 16;
    v16h afrag = frag_join(*(const v8h*)arow, *(const v8h*)(arow + 8));
#pragma unroll
    for (int nt = 0; nt < 4; ++nt) {
      const _Float16* brow = Wt + (nt * 16 + m) * 40 + g * 16;
      v16h bfrag = frag_join(*(const v8h*)brow, *(const v8h*)(brow + 8));
      WMMA(afrag, bfrag, acc[nt]);
    }
    __syncthreads();
  }

  if (TR) {
    // rows r are contiguous in [B,H,DK,S]: packed 128-bit stores
#pragma unroll
    for (int nt = 0; nt < 4; ++nt) {
      v8h pk;
#pragma unroll
      for (int r = 0; r < 8; ++r) pk[r] = (_Float16)acc[nt][r];
      const int col = nt * 16 + m;
      *(v8h*)(Out + ((size_t)bh * DK_ + col) * S_ + sBase + wv * 16 + g * 8) = pk;
    }
  } else {
    // transpose 16x64 tile through LDS -> coalesced b128 stores
    _Float16* Ep = lds + wv * (16 * 72);      // per-wave region, stride 72
#pragma unroll
    for (int nt = 0; nt < 4; ++nt)
#pragma unroll
      for (int r = 0; r < 8; ++r)
        Ep[(g * 8 + r) * 72 + nt * 16 + m] = (_Float16)acc[nt][r];
    __asm__ volatile("s_wait_dscnt 0" ::: "memory");
    const int lr = lane >> 1, c0 = (lane & 1) * 32;
    const _Float16* src = Ep + lr * 72 + c0;
    _Float16* dst = Out + ((size_t)bh * S_ + sBase + wv * 16 + lr) * DK_ + c0;
#pragma unroll
    for (int q = 0; q < 4; ++q)
      *(v8h*)(dst + q * 8) = *(const v8h*)(src + q * 8);
  }
}

// ---------------------------------------------------------------------------
// Kernel 2: causal flash attention. One wave per 16-query tile; 32-key tiles.
// qh,kh: [B,H,S,DK] f16.  vhT: [B,H,DK,S] f16.  ao: [B,S,H*DK] f16.
// ---------------------------------------------------------------------------
__global__ __launch_bounds__(128)
void attn_kernel(const _Float16* __restrict__ qh, const _Float16* __restrict__ kh,
                 const _Float16* __restrict__ vhT, _Float16* __restrict__ ao)
{
  const int b = blockIdx.z, h = blockIdx.y;
  const int tid = threadIdx.x, lane = tid & 31, wv = tid >> 5;
  const int qt = blockIdx.x * 4 + wv;      // 0..S/16-1
  const int qBase = qt * 16;
  const int bh = b * H_ + h;
  const int m = lane & 15, g = lane >> 4;

  __shared__ _Float16 Pbuf[4][16 * 72];    // per-wave scratch (P transpose + epilogue)
  _Float16* P = Pbuf[wv];

  const _Float16* Qrow = qh + ((size_t)bh * S_ + qBase + m) * DK_;
  v16h qf[2];
#pragma unroll
  for (int c = 0; c < 2; ++c) {
    const _Float16* p = Qrow + c * 32 + g * 16;
    qf[c] = frag_join(*(const v8h*)p, *(const v8h*)(p + 8));
  }

  v8f O[4];
#pragma unroll
  for (int nt = 0; nt < 4; ++nt) O[nt] = zero8();
  float mrun[8], lrun[8];
#pragma unroll
  for (int r = 0; r < 8; ++r) { mrun[r] = -1e30f; lrun[r] = 0.f; }

  const _Float16* Kbase = kh + (size_t)bh * S_ * DK_;
  const int nT = (qBase + 16 + 31) >> 5;   // causal: tiles up to the diagonal

  for (int t = 0; t < nT; ++t) {
    const int kb = t * 32;
    v8f s0 = zero8(), s1 = zero8();
#pragma unroll
    for (int c = 0; c < 2; ++c) {          // DK chunks of 32
      const _Float16* k0 = Kbase + (size_t)(kb + m) * DK_ + c * 32 + g * 16;
      v16h bf0 = frag_join(*(const v8h*)k0, *(const v8h*)(k0 + 8));
      WMMA(qf[c], bf0, s0);
      const _Float16* k1 = Kbase + (size_t)(kb + 16 + m) * DK_ + c * 32 + g * 16;
      v16h bf1 = frag_join(*(const v8h*)k1, *(const v8h*)(k1 + 8));
      WMMA(qf[c], bf1, s1);
    }

    // online softmax; C layout: element r of lane (g,m) is (row g*8+r, key m/m+16)
    const int key0 = kb + m, key1 = kb + 16 + m;
#pragma unroll
    for (int r = 0; r < 8; ++r) {
      const int qrow = qBase + g * 8 + r;
      float a0 = s0[r] * 0.125f; if (key0 > qrow) a0 = -1e30f;  // 1/sqrt(64)
      float a1 = s1[r] * 0.125f; if (key1 > qrow) a1 = -1e30f;
      float mt = fmaxf(a0, a1);
#pragma unroll
      for (int d = 1; d < 16; d <<= 1) mt = fmaxf(mt, __shfl_xor(mt, d, 32));
      const float mn = fmaxf(mrun[r], mt);
      const float alpha = __expf(mrun[r] - mn);
      const float p0 = __expf(a0 - mn), p1 = __expf(a1 - mn);
      float rs = p0 + p1;
#pragma unroll
      for (int d = 1; d < 16; d <<= 1) rs += __shfl_xor(rs, d, 32);
      lrun[r] = lrun[r] * alpha + rs;
      mrun[r] = mn;
#pragma unroll
      for (int nt = 0; nt < 4; ++nt) O[nt][r] *= alpha;
      P[(g * 8 + r) * 72 + m]      = (_Float16)p0;   // transpose via LDS
      P[(g * 8 + r) * 72 + 16 + m] = (_Float16)p1;
    }
    __asm__ volatile("s_wait_dscnt 0" ::: "memory");

    const _Float16* prow = P + m * 72 + g * 16;
    v16h pf = frag_join(*(const v8h*)prow, *(const v8h*)(prow + 8));
#pragma unroll
    for (int nt = 0; nt < 4; ++nt) {
      const _Float16* vp = vhT + ((size_t)bh * DK_ + nt * 16 + m) * S_ + kb + g * 16;
      v16h vf = frag_join(*(const v8h*)vp, *(const v8h*)(vp + 8));
      WMMA(pf, vf, O[nt]);
    }
  }

  // normalize, transpose 16x64 tile through LDS, coalesced b128 stores
#pragma unroll
  for (int nt = 0; nt < 4; ++nt)
#pragma unroll
    for (int r = 0; r < 8; ++r)
      P[(g * 8 + r) * 72 + nt * 16 + m] = (_Float16)(O[nt][r] / lrun[r]);
  __asm__ volatile("s_wait_dscnt 0" ::: "memory");
  {
    const int lr = lane >> 1, c0 = (lane & 1) * 32;
    const _Float16* src = P + lr * 72 + c0;
    _Float16* dst = ao + ((size_t)b * S_ + qBase + lr) * (H_ * DK_) + h * DK_ + c0;
#pragma unroll
    for (int q = 0; q < 4; ++q)
      *(v8h*)(dst + q * 8) = *(const v8h*)(src + q * 8);
  }
}

// ---------------------------------------------------------------------------
// Kernel 3: output projection  out[bs, n] = ao[bs, :] @ Wo[n, :] + bo[n]  (f32)
// All 4 waves share the same Wo column tile -> stage it once in LDS.
// ---------------------------------------------------------------------------
__global__ __launch_bounds__(128)
void outproj_kernel(const _Float16* __restrict__ ao, const float* __restrict__ Wo,
                    const float* __restrict__ bo, float* __restrict__ out)
{
  const int tid = threadIdx.x, lane = tid & 31, wv = tid >> 5;
  const int rt = blockIdx.x * 4 + wv;      // 16-row tile over B*S
  const int colBase = blockIdx.y * 64;
  const int m = lane & 15, g = lane >> 4;

  __shared__ _Float16 WoT[64 * 40];        // [n][k] f16 tile, stride 40

  v8f acc[4];
#pragma unroll
  for (int nt = 0; nt < 4; ++nt) acc[nt] = zero8();

  const _Float16* arow = ao + (size_t)(rt * 16 + m) * (H_ * DK_);

  for (int kb = 0; kb < H_ * DK_; kb += 32) {
    {
      const int n = tid >> 1;              // 0..63
      const int ksel = (tid & 1) * 16;
      const float* src = Wo + (size_t)(colBase + n) * D_ + kb + ksel;
      v8h wa, wb;
#pragma unroll
      for (int j = 0; j < 8; ++j) wa[j] = (_Float16)src[j];
#pragma unroll
      for (int j = 0; j < 8; ++j) wb[j] = (_Float16)src[8 + j];
      *(v8h*)(WoT + n * 40 + ksel)     = wa;
      *(v8h*)(WoT + n * 40 + ksel + 8) = wb;
    }
    __syncthreads();

    const _Float16* ap = arow + kb + g * 16;
    v16h af = frag_join(*(const v8h*)ap, *(const v8h*)(ap + 8));
#pragma unroll
    for (int nt = 0; nt < 4; ++nt) {
      const _Float16* brow = WoT + (nt * 16 + m) * 40 + g * 16;
      v16h bf = frag_join(*(const v8h*)brow, *(const v8h*)(brow + 8));
      WMMA(af, bf, acc[nt]);
    }
    __syncthreads();
  }

  // f32 scalar stores coalesce to full 64B sectors per half-wave
#pragma unroll
  for (int nt = 0; nt < 4; ++nt) {
    const int col = colBase + nt * 16 + m;
    const float bias = bo[col];
#pragma unroll
    for (int r = 0; r < 8; ++r) {
      const int row = rt * 16 + g * 8 + r;
      out[(size_t)row * D_ + col] = acc[nt][r] + bias;
    }
  }
}

// ---------------------------------------------------------------------------
extern "C" void kernel_launch(void* const* d_in, const int* in_sizes, int n_in,
                              void* d_out, int out_size, void* d_ws, size_t ws_size,
                              hipStream_t stream)
{
  (void)in_sizes; (void)n_in; (void)out_size; (void)ws_size;
  const float* query = (const float*)d_in[0];
  const float* key   = (const float*)d_in[1];
  const float* value = (const float*)d_in[2];
  const float* Wq    = (const float*)d_in[3];
  const float* Wk    = (const float*)d_in[4];
  const float* Wv    = (const float*)d_in[5];
  const float* Wo    = (const float*)d_in[6];
  const float* bo    = (const float*)d_in[7];
  float* out = (float*)d_out;

  const size_t headElems = (size_t)B_ * H_ * S_ * DK_;   // 4M halves = 8MB each
  _Float16* qh  = (_Float16*)d_ws;
  _Float16* kh  = qh  + headElems;
  _Float16* vhT = kh  + headElems;
  _Float16* ao  = vhT + headElems;                       // total 32MB of ws

  dim3 gp(S_ / 64, H_, B_), bp(128);
  proj_kernel<0><<<gp, bp, 0, stream>>>(query, Wq, qh);
  proj_kernel<0><<<gp, bp, 0, stream>>>(key,   Wk, kh);
  proj_kernel<1><<<gp, bp, 0, stream>>>(value, Wv, vhT);

  attn_kernel<<<dim3(S_ / 64, H_, B_), dim3(128), 0, stream>>>(qh, kh, vhT, ao);

  outproj_kernel<<<dim3((B_ * S_) / 64, D_ / 64), dim3(128), 0, stream>>>(
      ao, Wo, bo, out);
}
